// CosineCrossAttention_67903432950161
// MI455X (gfx1250) — matrix operand order repaired and checked by reading, not compiled, not run on hardware
//
#include <hip/hip_runtime.h>

// Problem constants (from reference): B=8,T=8,N=2048,D=384,H=8,Dh=48
#define Dd     384
#define Nn     2048
#define Hh     8
#define Dh     48
#define TILE_M 128
#define PITCH  392            // halves per LDS row (384 + 8 pad -> conflict-free ds_load_b128)
#define NBT    64             // B*T slabs

typedef __attribute__((ext_vector_type(16))) __bf16 bf16x16;
typedef __attribute__((ext_vector_type(8)))  float  f32x8;

union Frag { bf16x16 v; uint4 q[2]; };

__device__ __forceinline__ unsigned short f2bf(float x) {
    unsigned int u = __float_as_uint(x);
    u += 0x7FFFu + ((u >> 16) & 1u);          // round-to-nearest-even
    return (unsigned short)(u >> 16);
}

// ---------------------------------------------------------------------------
// Prep: Wq, Wp -> transposed bf16 copies so WMMA B-fragments load contiguously
// ---------------------------------------------------------------------------
__global__ void prep_w(const float* __restrict__ Wq, const float* __restrict__ Wp,
                       unsigned short* __restrict__ WqT, unsigned short* __restrict__ WpT) {
    int idx = blockIdx.x * 256 + threadIdx.x;
    if (idx >= Dd * Dd) return;
    int k = idx / Dd, n = idx - k * Dd;
    WqT[n * Dd + k] = f2bf(Wq[idx]);
    WpT[n * Dd + k] = f2bf(Wp[idx]);
}

// ---------------------------------------------------------------------------
// kv projections: kproj[bt][:] = kv@Wk (f32), vkp[bt][dh*8+h] = (kv@Wv)/|k_h|
// (output permute d = dh*H + h and the 1/|k| factor are folded in here)
// ---------------------------------------------------------------------------
__global__ void kv_proj(const float* __restrict__ kv, const float* __restrict__ Wk,
                        const float* __restrict__ Wv, float* __restrict__ kproj,
                        float* __restrict__ vkp) {
    __shared__ float skv[Dd];
    __shared__ float sk[Dd];
    __shared__ float knrm[Hh];
    const int bt = blockIdx.x, j = threadIdx.x;
    skv[j] = kv[bt * Dd + j];
    __syncthreads();
    float kj = 0.f, vj = 0.f;
    for (int i = 0; i < Dd; ++i) {
        float x = skv[i];
        kj += x * Wk[i * Dd + j];
        vj += x * Wv[i * Dd + j];
    }
    kproj[bt * Dd + j] = kj;
    sk[j] = kj * kj;
    __syncthreads();
    if (j < Hh) {
        float s = 0.f;
        for (int d = 0; d < Dh; ++d) s += sk[j * Dh + d];
        knrm[j] = sqrtf(s);
    }
    __syncthreads();
    int h = j / Dh, dh = j - h * Dh;
    vkp[bt * Dd + dh * Hh + h] = vj / knrm[h];
}

// ---------------------------------------------------------------------------
// Fused: q=A@Wq (WMMA, reductions only) -> attn scalars -> X -> X@Wp + bp
// ---------------------------------------------------------------------------
__global__ __launch_bounds__(256) void fused_attn(
        const float* __restrict__ query, const float* __restrict__ kproj,
        const float* __restrict__ vkp, const unsigned short* __restrict__ WqT,
        const unsigned short* __restrict__ WpT, const float* __restrict__ bias,
        float* __restrict__ out) {
    __shared__ __align__(16) unsigned short tile[TILE_M * PITCH]; // ~98 KB (A, then X)
    __shared__ float qn2[TILE_M * Hh];
    __shared__ float qdk[TILE_M * Hh];

    const int tid  = threadIdx.x;
    const int wave = tid >> 5, lane = tid & 31, l15 = lane & 15, hf = lane >> 4;
    const int bt   = blockIdx.x >> 4;
    const int trow = (blockIdx.x & 15) * TILE_M;
    const long rowbase = (long)bt * Nn + trow;
    const int m0 = wave * 16;

    for (int i = tid; i < TILE_M * Hh; i += 256) { qn2[i] = 0.f; qdk[i] = 0.f; }

    // ---- stage query tile as bf16 in LDS (f32x4 loads -> packed bf16 stores)
    {
        const float4* src = (const float4*)(query + rowbase * Dd);
        for (int i = tid; i < TILE_M * Dd / 4; i += 256) {
            int e = i * 4, m = e / Dd, d = e - m * Dd;
            float4 f = src[i];
            unsigned int lo = (unsigned)f2bf(f.x) | ((unsigned)f2bf(f.y) << 16);
            unsigned int hi = (unsigned)f2bf(f.z) | ((unsigned)f2bf(f.w) << 16);
            *(uint2*)&tile[m * PITCH + d] = make_uint2(lo, hi);
        }
    }
    __syncthreads();

    // ---- GEMM1: q = tile @ Wq ; consume q only through per-(row,head) sums.
    //      Each head spans 3 column tiles (48 = 3*16): accumulate per-lane
    //      partials across the head, then do one butterfly + atomic per head.
    for (int h = 0; h < Hh; ++h) {
        float s2l[8], dkl[8];
#pragma unroll
        for (int r = 0; r < 8; ++r) { s2l[r] = 0.f; dkl[r] = 0.f; }

        for (int t3 = 0; t3 < 3; ++t3) {
            const int n0 = (h * 3 + t3) * 16;
            const float kval = kproj[bt * Dd + n0 + l15];  // k for this lane's column
            f32x8 acc = {0.f, 0.f, 0.f, 0.f, 0.f, 0.f, 0.f, 0.f};
            const unsigned short* brow = WqT + (n0 + l15) * Dd;
            for (int kk = 0; kk < Dd; kk += 32) {
                Frag a, b;
                const uint4* ap = (const uint4*)(&tile[(m0 + l15) * PITCH + kk + hf * 16]);
                a.q[0] = ap[0];           // K = kk + hf*16      .. +7
                a.q[1] = ap[1];           // K = kk + hf*16 + 8  .. +15
                const uint4* bp2 = (const uint4*)(brow + kk + hf * 16);
                b.q[0] = bp2[0];          // K = kk + hf*16      .. +7
                b.q[1] = bp2[1];          // K = kk + hf*16 + 8  .. +15
                acc = __builtin_amdgcn_wmma_f32_16x16x32_bf16(
                          false, a.v, false, b.v, (short)0, acc, false, false);
            }
#pragma unroll
            for (int r = 0; r < 8; ++r) {
                float v = acc[r];
                s2l[r] += v * v;
                dkl[r] += v * kval;
            }
        }
        // one cross-lane reduction + LDS atomic per (row, head)
#pragma unroll
        for (int r = 0; r < 8; ++r) {
            float s2 = s2l[r], dk = dkl[r];
            for (int m = 1; m <= 8; m <<= 1) {
                s2 += __shfl_xor(s2, m, 32);
                dk += __shfl_xor(dk, m, 32);
            }
            if (l15 == 0) {
                int row = m0 + r + hf * 8;
                atomicAdd(&qn2[row * Hh + h], s2);
                atomicAdd(&qdk[row * Hh + h], dk);
            }
        }
    }
    __syncthreads();

    // ---- attn scalar per (row, head): qdk / |q|  (1/|k| already in vkp)
    for (int i = tid; i < TILE_M * Hh; i += 256)
        qdk[i] = qdk[i] * rsqrtf(qn2[i]);
    __syncthreads();

    // ---- rebuild X in-place: X[m][dh*8+h] = attn[m][h] * vkp[bt][dh*8+h]
    {
        const int m = tid >> 1;
        const int d0 = (tid & 1) * (Dd / 2);
        const float* vk = vkp + bt * Dd;
        for (int d = d0; d < d0 + Dd / 2; ++d)
            tile[m * PITCH + d] = f2bf(qdk[m * Hh + (d & 7)] * vk[d]);
    }
    __syncthreads();

    // ---- GEMM2: out = X @ Wp + bp
    for (int nt = 0; nt < Dd / 16; ++nt) {
        const int n0 = nt * 16;
        f32x8 acc = {0.f, 0.f, 0.f, 0.f, 0.f, 0.f, 0.f, 0.f};
        const unsigned short* brow = WpT + (n0 + l15) * Dd;
        for (int kk = 0; kk < Dd; kk += 32) {
            Frag a, b;
            const uint4* ap = (const uint4*)(&tile[(m0 + l15) * PITCH + kk + hf * 16]);
            a.q[0] = ap[0];
            a.q[1] = ap[1];
            const uint4* bp2 = (const uint4*)(brow + kk + hf * 16);
            b.q[0] = bp2[0];
            b.q[1] = bp2[1];
            acc = __builtin_amdgcn_wmma_f32_16x16x32_bf16(
                      false, a.v, false, b.v, (short)0, acc, false, false);
        }
        const float bv = bias[n0 + l15];
        const int col = n0 + l15;
#pragma unroll
        for (int r = 0; r < 8; ++r) {
            long row = rowbase + m0 + r + hf * 8;
            out[row * Dd + col] = acc[r] + bv;
        }
    }
}

// ---------------------------------------------------------------------------
extern "C" void kernel_launch(void* const* d_in, const int* in_sizes, int n_in,
                              void* d_out, int out_size, void* d_ws, size_t ws_size,
                              hipStream_t stream) {
    (void)in_sizes; (void)n_in; (void)out_size; (void)ws_size;
    const float* query = (const float*)d_in[0];
    const float* kv    = (const float*)d_in[1];
    const float* Wq    = (const float*)d_in[2];
    const float* Wk    = (const float*)d_in[3];
    const float* Wv    = (const float*)d_in[4];
    const float* Wp    = (const float*)d_in[5];
    const float* bp    = (const float*)d_in[6];
    float* out = (float*)d_out;

    char* ws = (char*)d_ws;
    unsigned short* WqT  = (unsigned short*)(ws);                      // 294912 B
    unsigned short* WpT  = (unsigned short*)(ws + 294912);             // 294912 B
    float*          kprj = (float*)(ws + 2 * 294912);                  //  98304 B
    float*          vkp  = (float*)(ws + 2 * 294912 + 98304);          //  98304 B

    hipLaunchKernelGGL(prep_w, dim3((Dd * Dd + 255) / 256), dim3(256), 0, stream,
                       Wq, Wp, WqT, WpT);
    hipLaunchKernelGGL(kv_proj, dim3(NBT), dim3(Dd), 0, stream,
                       kv, Wk, Wv, kprj, vkp);
    hipLaunchKernelGGL(fused_attn, dim3(NBT * (Nn / TILE_M)), dim3(256), 0, stream,
                       query, kprj, vkp, WqT, WpT, bp, out);
}
